// Hyper_3453153706530
// MI455X (gfx1250) — matrix operand order repaired and checked by reading, not compiled, not run on hardware
//
#include <hip/hip_runtime.h>
#include <hip/hip_bf16.h>

typedef _Float16 half_t;
typedef __attribute__((ext_vector_type(16))) _Float16 v16h;
typedef __attribute__((ext_vector_type(8)))  _Float16 v8h;
typedef __attribute__((ext_vector_type(8)))  float    v8f;
typedef __attribute__((ext_vector_type(4)))  float    v4f;

#define BB    1024
#define NN    100000
#define DD    200
#define OC    32
#define FW    9
#define LL    192          // D - FW + 1
#define OCFW  288          // OC*FW
#define KH    6144         // OC*L  (K of h-GEMM)
#define DPAD  208          // D padded to 16 for h-GEMM N dim
#define HPAD  224          // D padded to 32 for logits-GEMM K dim
#define EPS   1e-5f

// ---- workspace byte offsets ----
#define WS_X    0u                    // x      : 1024*200 f32      = 819200 B
#define WS_KF   819200u               // kfilt  : 1024*288 f32      = 1179648 B
#define WS_Y16  1998848u              // y f16  : 1024*6144 f16     = 12582912 B
#define WS_W16T 14581760u             // W_fc^T : 208*6144 f16      = 2555904 B
#define WS_H16  17137664u             // h f16  : 1024*224 f16      = 458752 B

// ---------------- glue kernels ----------------

__global__ void k_gather_bn0(const int* __restrict__ e1_idx,
                             const float* __restrict__ E,
                             const float* __restrict__ g, const float* __restrict__ be,
                             const float* __restrict__ mu, const float* __restrict__ va,
                             float* __restrict__ x) {
  int tid = blockIdx.x * blockDim.x + threadIdx.x;
  if (tid >= BB * DD) return;
  int b = tid / DD, d = tid - b * DD;
  float s = g[0] * rsqrtf(va[0] + EPS);
  x[tid] = (E[(size_t)e1_idx[b] * DD + d] - mu[0]) * s + be[0];
}

__global__ void k_kfilter(const int* __restrict__ r_idx,
                          const float* __restrict__ R,
                          const float* __restrict__ W_fc1,
                          const float* __restrict__ b_fc1,
                          float* __restrict__ kf) {
  int tid = blockIdx.x * blockDim.x + threadIdx.x;
  if (tid >= BB * OCFW) return;
  int b = tid / OCFW, j = tid - b * OCFW;
  const float* rr = R + (size_t)r_idx[b] * DD;
  float acc = b_fc1[j];
  #pragma unroll 8
  for (int d = 0; d < DD; ++d) acc += rr[d] * W_fc1[(size_t)d * OCFW + j];
  kf[tid] = acc;
}

// W_fc (6144 x 200) -> transposed padded f16 (208 x 6144): w16t[n*6144 + k]
__global__ void k_wfc_to_f16t(const float* __restrict__ W_fc, half_t* __restrict__ w16t) {
  int tid = blockIdx.x * blockDim.x + threadIdx.x;
  if (tid >= DPAD * KH) return;
  int n = tid / KH, k = tid - n * KH;
  w16t[tid] = (n < DD) ? (half_t)W_fc[(size_t)k * DD + n] : (half_t)0.0f;
}

__global__ void k_h16_zero(half_t* __restrict__ h16) {
  int tid = blockIdx.x * blockDim.x + threadIdx.x;
  if (tid < BB * HPAD) h16[tid] = (half_t)0.0f;
}

// per-sample conv (FW=9) + BN1, store y as f16 row-major (1024 x 6144)
__global__ void k_conv_bn1(const float* __restrict__ x,
                           const float* __restrict__ kf,
                           const float* __restrict__ g, const float* __restrict__ be,
                           const float* __restrict__ mu, const float* __restrict__ va,
                           half_t* __restrict__ y16) {
  int tid = blockIdx.x * blockDim.x + threadIdx.x;
  if (tid >= BB * OC * LL) return;
  int b = tid / (OC * LL);
  int rem = tid - b * (OC * LL);
  int o = rem / LL, l = rem - o * LL;
  const float* xr = x + (size_t)b * DD + l;
  const float* kr = kf + (size_t)b * OCFW + o * FW;
  float acc = 0.f;
  #pragma unroll
  for (int w = 0; w < FW; ++w) acc += xr[w] * kr[w];
  float s = g[o] * rsqrtf(va[o] + EPS);
  acc = (acc - mu[o]) * s + be[o];
  y16[tid] = (half_t)acc;
}

// ---------------- WMMA GEMM 1: h = y @ W_fc (+bias, BN2, ReLU) ----------------
// one 16x16 tile per wave; A = y16 (1024x6144), B = w16t (col-major rows), K=6144
__global__ void k_gemm_h(const half_t* __restrict__ y16,
                         const half_t* __restrict__ w16t,
                         const float* __restrict__ b_fc,
                         const float* __restrict__ g, const float* __restrict__ be,
                         const float* __restrict__ mu, const float* __restrict__ va,
                         half_t* __restrict__ h16) {
  int wid    = threadIdx.x >> 5;
  int lane   = threadIdx.x & 31;
  int lane16 = lane & 15;
  int khalf  = lane >> 4;
  int tile   = blockIdx.x * 8 + wid;          // 104 blocks * 8 waves = 832 tiles
  int mtile  = tile / 13;                     // 64 m-tiles
  int ntile  = tile - mtile * 13;             // 13 n-tiles (208 cols)

  int m = mtile * 16 + lane16;
  int n = ntile * 16 + lane16;
  const half_t* arow = y16 + (size_t)m * KH;
  const half_t* bcol = w16t + (size_t)n * KH;

  v8f acc = {};
  for (int kk = 0; kk < KH / 32; ++kk) {
    int abase = kk * 32 + khalf * 16;
    v8h alo = *(const v8h*)(arow + abase);
    v8h ahi = *(const v8h*)(arow + abase + 8);
    v16h a = __builtin_shufflevector(alo, ahi, 0,1,2,3,4,5,6,7,8,9,10,11,12,13,14,15);
    int bbase = kk * 32 + khalf * 16;
    v8h blo = *(const v8h*)(bcol + bbase);
    v8h bhi = *(const v8h*)(bcol + bbase + 8);
    v16h bf = __builtin_shufflevector(blo, bhi, 0,1,2,3,4,5,6,7,8,9,10,11,12,13,14,15);
    acc = __builtin_amdgcn_wmma_f32_16x16x32_f16(false, a, false, bf, (short)0, acc, false, false);
  }

  // epilogue: C/D layout -> (m = v + 8*khalf, n = lane16)
  #pragma unroll
  for (int v = 0; v < 8; ++v) {
    int mm = mtile * 16 + v + 8 * khalf;
    int nn = ntile * 16 + lane16;
    if (nn < DD) {
      float val = acc[v] + b_fc[nn];
      val = (val - mu[nn]) * (g[nn] * rsqrtf(va[nn] + EPS)) + be[nn];
      val = val > 0.f ? val : 0.f;
      h16[(size_t)mm * HPAD + nn] = (half_t)val;
    }
  }
}

// ---------------- WMMA GEMM 2: logits = h @ E^T, sigmoid ----------------
// block = 256 thr (8 waves). blockIdx = n-tile (16 E rows, streamed once).
// wave `wid` covers m rows [wid*128, wid*128+128) with 8 accumulators.
__global__ void __launch_bounds__(256)
k_gemm_logits(const half_t* __restrict__ h16,
              const float* __restrict__ E,
              const float* __restrict__ bvec,
              float* __restrict__ out) {
  int wid    = threadIdx.x >> 5;
  int lane   = threadIdx.x & 31;
  int lane16 = lane & 15;
  int khalf  = lane >> 4;
  int ntile  = blockIdx.x;                    // 6250 tiles
  int mblock = wid * 128;

  int n = ntile * 16 + lane16;                // entity row of E
  const float* erow = E + (size_t)n * DD;

  v8f acc[8];
  #pragma unroll
  for (int i = 0; i < 8; ++i) acc[i] = (v8f){};

  #pragma unroll 1
  for (int kk = 0; kk < HPAD / 32; ++kk) {    // 7 K-steps over padded K=224
    int kbase = kk * 32 + khalf * 16;
    v16h bf;
    if (kbase + 16 <= DD) {
      const v4f* p = (const v4f*)(erow + kbase);
      #pragma unroll
      for (int q = 0; q < 4; ++q) {
        v4f f = p[q];
        bf[4*q+0] = (half_t)f.x; bf[4*q+1] = (half_t)f.y;
        bf[4*q+2] = (half_t)f.z; bf[4*q+3] = (half_t)f.w;
      }
    } else {
      #pragma unroll
      for (int j = 0; j < 16; ++j) {
        int k = kbase + j;
        bf[j] = (k < DD) ? (half_t)erow[k] : (half_t)0.0f;
      }
    }
    int abase = kk * 32 + khalf * 16;
    #pragma unroll
    for (int mi = 0; mi < 8; ++mi) {
      const half_t* arow = h16 + (size_t)(mblock + mi * 16 + lane16) * HPAD + abase;
      v8h alo = *(const v8h*)(arow);
      v8h ahi = *(const v8h*)(arow + 8);
      v16h a = __builtin_shufflevector(alo, ahi, 0,1,2,3,4,5,6,7,8,9,10,11,12,13,14,15);
      acc[mi] = __builtin_amdgcn_wmma_f32_16x16x32_f16(false, a, false, bf, (short)0, acc[mi], false, false);
    }
  }

  #pragma unroll
  for (int mi = 0; mi < 8; ++mi) {
    #pragma unroll
    for (int v = 0; v < 8; ++v) {
      int mm = mblock + mi * 16 + v + 8 * khalf;
      int nn = ntile * 16 + lane16;
      float lg = acc[mi][v] + bvec[nn];
      out[(size_t)mm * NN + nn] = 1.0f / (1.0f + expf(-lg));
    }
  }
}

// ---------------- launcher ----------------
extern "C" void kernel_launch(void* const* d_in, const int* in_sizes, int n_in,
                              void* d_out, int out_size, void* d_ws, size_t ws_size,
                              hipStream_t stream) {
  const int*   e1_idx = (const int*)  d_in[0];
  const int*   r_idx  = (const int*)  d_in[1];
  const float* E      = (const float*)d_in[2];
  const float* R      = (const float*)d_in[3];
  const float* W_fc   = (const float*)d_in[4];
  const float* b_fc   = (const float*)d_in[5];
  const float* W_fc1  = (const float*)d_in[6];
  const float* b_fc1  = (const float*)d_in[7];
  const float* bn0_g  = (const float*)d_in[8];
  const float* bn0_b  = (const float*)d_in[9];
  const float* bn0_m  = (const float*)d_in[10];
  const float* bn0_v  = (const float*)d_in[11];
  const float* bn1_g  = (const float*)d_in[12];
  const float* bn1_b  = (const float*)d_in[13];
  const float* bn1_m  = (const float*)d_in[14];
  const float* bn1_v  = (const float*)d_in[15];
  const float* bn2_g  = (const float*)d_in[16];
  const float* bn2_b  = (const float*)d_in[17];
  const float* bn2_m  = (const float*)d_in[18];
  const float* bn2_v  = (const float*)d_in[19];
  const float* bvec   = (const float*)d_in[20];
  float* out = (float*)d_out;

  char* ws = (char*)d_ws;
  float*  x    = (float*) (ws + WS_X);
  float*  kf   = (float*) (ws + WS_KF);
  half_t* y16  = (half_t*)(ws + WS_Y16);
  half_t* w16t = (half_t*)(ws + WS_W16T);
  half_t* h16  = (half_t*)(ws + WS_H16);

  k_gather_bn0<<<(BB*DD + 255)/256, 256, 0, stream>>>(e1_idx, E, bn0_g, bn0_b, bn0_m, bn0_v, x);
  k_kfilter  <<<(BB*OCFW + 255)/256, 256, 0, stream>>>(r_idx, R, W_fc1, b_fc1, kf);
  k_wfc_to_f16t<<<(DPAD*KH + 255)/256, 256, 0, stream>>>(W_fc, w16t);
  k_h16_zero <<<(BB*HPAD + 255)/256, 256, 0, stream>>>(h16);
  k_conv_bn1 <<<(BB*OC*LL + 255)/256, 256, 0, stream>>>(x, kf, bn1_g, bn1_b, bn1_m, bn1_v, y16);
  k_gemm_h   <<<104, 256, 0, stream>>>(y16, w16t, b_fc, bn2_g, bn2_b, bn2_m, bn2_v, h16);
  k_gemm_logits<<<NN/16, 256, 0, stream>>>(h16, E, bvec, out);
}
